// GraphGAT3LayerClassifierStats_1949915152979
// MI455X (gfx1250) — matrix operand order repaired and checked by reading, not compiled, not run on hardware
//
#include <hip/hip_runtime.h>
#include <hip/hip_bf16.h>

typedef __attribute__((ext_vector_type(16))) __bf16 v16bf;
typedef __attribute__((ext_vector_type(8)))  __bf16 v8bf;
typedef __attribute__((ext_vector_type(8)))  float  v8f;

#define NNODE 20000
#define NEDGE 320000
#define NG    64

// ---------------- generic fill ----------------
__global__ void fill_k(float* __restrict__ p, float v, long n) {
    long t = (long)blockIdx.x * blockDim.x + threadIdx.x;
    if (t < n) p[t] = v;
}

// ---------------- self-loop edge-attr stats ----------------
__global__ void edge_stats_k(const int* __restrict__ dst, const float* __restrict__ ea,
                             float* __restrict__ cnt, float* __restrict__ sumea, int E) {
    int t = blockIdx.x * blockDim.x + threadIdx.x;
    if (t >= E * 16) return;
    int e = t >> 4, d = t & 15;
    int dd = dst[e];
    atomicAdd(&sumea[dd * 16 + d], ea[t]);
    if (d == 0) atomicAdd(&cnt[dd], 1.0f);
}

__global__ void loop_ea_k(const float* __restrict__ cnt, float* __restrict__ sumea, int N) {
    int t = blockIdx.x * blockDim.x + threadIdx.x;
    if (t >= N * 16) return;
    int n = t >> 4;
    sumea[t] = sumea[t] / fmaxf(cnt[n], 1.0f);
}

// ---------------- graph-feature mean pool (x[:,64:80]) ----------------
__global__ void gstat_k(const float* __restrict__ x, const int* __restrict__ batch,
                        float* __restrict__ gsum, float* __restrict__ gcnt, int N) {
    int t = blockIdx.x * blockDim.x + threadIdx.x;
    if (t >= N * 16) return;
    int n = t >> 4, d = t & 15;
    int b = batch[n];
    atomicAdd(&gsum[b * 16 + d], x[n * 80 + 64 + d]);
    if (d == 0) atomicAdd(&gcnt[b], 1.0f);
}

__global__ void gdiv_k(const float* __restrict__ gsum, const float* __restrict__ gcnt,
                       float* __restrict__ out, int G, int F) {
    int t = blockIdx.x * blockDim.x + threadIdx.x;
    if (t >= G * F) return;
    int g = t / F;
    out[t] = gsum[t] / fmaxf(gcnt[g], 1.0f);
}

// ---------------- bf16 packing kernels ----------------
// pack node part of x (N x 80, cols 0..63) into dense bf16 N x 64
__global__ void pack_x_bf16_k(const float* __restrict__ x, __bf16* __restrict__ xbf, int N) {
    int t = blockIdx.x * blockDim.x + threadIdx.x;
    if (t >= N * 64) return;
    int n = t >> 6, c = t & 63;
    xbf[t] = (__bf16)x[n * 80 + c];
}

// W (K x 256 f32 row-major) -> Wt (256 x K bf16 row-major), i.e. transposed
__global__ void wt_bf16_k(const float* __restrict__ W, __bf16* __restrict__ Wt, int K) {
    int t = blockIdx.x * blockDim.x + threadIdx.x;
    if (t >= K * 256) return;
    int c = t / K, k = t - c * K;
    Wt[t] = (__bf16)W[k * 256 + c];
}

// ---------------- bf16 WMMA GEMM: C[N x 256] = A[N x K] @ Wt^T ----------------
// A: bf16 N x K row-major. Wt: bf16 256 x K (pre-transposed weights).
// One wave32 computes a 16x64 strip: 4 accumulators share one A fragment.
// Fragment loads are contiguous b128s thanks to bf16 pre-pack + weight transpose.
__global__ void gemm_bf16t_wmma(const __bf16* __restrict__ A, const __bf16* __restrict__ Wt,
                                float* __restrict__ C, int Nrows, int K) {
    const int lane = threadIdx.x & 31;
    const int wave = threadIdx.x >> 5;
    const int wt   = blockIdx.x * (blockDim.x >> 5) + wave;
    if (wt >= (Nrows / 16) * 4) return;          // grid sized to full waves
    const int tr = wt >> 2, cg = wt & 3;         // row tile, 64-col group
    const int half = lane >> 4, mr = lane & 15;
    const __bf16* __restrict__ arow = A + (long)(tr * 16 + mr) * K;
    const __bf16* __restrict__ bc0 = Wt + (long)(cg * 64 +  0 + mr) * K;
    const __bf16* __restrict__ bc1 = Wt + (long)(cg * 64 + 16 + mr) * K;
    const __bf16* __restrict__ bc2 = Wt + (long)(cg * 64 + 32 + mr) * K;
    const __bf16* __restrict__ bc3 = Wt + (long)(cg * 64 + 48 + mr) * K;
    v8f acc0 = {}, acc1 = {}, acc2 = {}, acc3 = {};
    for (int k0 = 0; k0 < K; k0 += 32) {
        __builtin_prefetch(arow + k0 + 64, 0, 1);
        __builtin_prefetch(bc0 + k0 + 64, 0, 1);
        // A 16x32 bf16 layout: lane covers K = k0+16h..+15 (two b128s)
        v8bf alo = *(const v8bf*)(arow + k0 + 16 * half);
        v8bf ahi = *(const v8bf*)(arow + k0 + 16 * half + 8);
        v16bf a = __builtin_shufflevector(alo, ahi, 0,1,2,3,4,5,6,7,8,9,10,11,12,13,14,15);
        // B 32x16 layout: lane covers K = k0+16h..+15 of its column (two b128s)
        const int kb = k0 + 16 * half;
        v8bf l0 = *(const v8bf*)(bc0 + kb), h0 = *(const v8bf*)(bc0 + kb + 8);
        v16bf b0 = __builtin_shufflevector(l0, h0, 0,1,2,3,4,5,6,7,8,9,10,11,12,13,14,15);
        acc0 = __builtin_amdgcn_wmma_f32_16x16x32_bf16(false, a, false, b0, (short)0, acc0, false, false);
        v8bf l1 = *(const v8bf*)(bc1 + kb), h1 = *(const v8bf*)(bc1 + kb + 8);
        v16bf b1 = __builtin_shufflevector(l1, h1, 0,1,2,3,4,5,6,7,8,9,10,11,12,13,14,15);
        acc1 = __builtin_amdgcn_wmma_f32_16x16x32_bf16(false, a, false, b1, (short)0, acc1, false, false);
        v8bf l2 = *(const v8bf*)(bc2 + kb), h2 = *(const v8bf*)(bc2 + kb + 8);
        v16bf b2 = __builtin_shufflevector(l2, h2, 0,1,2,3,4,5,6,7,8,9,10,11,12,13,14,15);
        acc2 = __builtin_amdgcn_wmma_f32_16x16x32_bf16(false, a, false, b2, (short)0, acc2, false, false);
        v8bf l3 = *(const v8bf*)(bc3 + kb), h3 = *(const v8bf*)(bc3 + kb + 8);
        v16bf b3 = __builtin_shufflevector(l3, h3, 0,1,2,3,4,5,6,7,8,9,10,11,12,13,14,15);
        acc3 = __builtin_amdgcn_wmma_f32_16x16x32_bf16(false, a, false, b3, (short)0, acc3, false, false);
    }
#pragma unroll
    for (int r = 0; r < 8; ++r) {
        long base = (long)(tr * 16 + r + 8 * half) * 256 + cg * 64 + mr;
        C[base +  0] = acc0[r];
        C[base + 16] = acc1[r];
        C[base + 32] = acc2[r];
        C[base + 48] = acc3[r];
    }
}

// ---------------- per-node attention logits ----------------
__global__ void node_alpha_k(const float* __restrict__ h, const float* __restrict__ a_s,
                             const float* __restrict__ a_d, float* __restrict__ als,
                             float* __restrict__ ald, int N) {
    int t = blockIdx.x * blockDim.x + threadIdx.x;
    if (t >= N * 4) return;
    int n = t >> 2, hh = t & 3;
    const float* hp = h + (long)n * 256 + hh * 64;
    const float* sp = a_s + hh * 64;
    const float* dp = a_d + hh * 64;
    float ss = 0.f, sd = 0.f;
#pragma unroll 8
    for (int c = 0; c < 64; ++c) { float v = hp[c]; ss += v * sp[c]; sd += v * dp[c]; }
    als[t] = ss; ald[t] = sd;
}

// ---------------- folded edge projection: v[d][h] = sum_c We[d, h*64+c]*ae[h,c] ----------------
__global__ void edge_vec_k(const float* __restrict__ We, const float* __restrict__ ae,
                           float* __restrict__ v) {
    int t = threadIdx.x;
    if (t >= 64) return;
    int d = t >> 2, hh = t & 3;
    float s = 0.f;
    for (int c = 0; c < 64; ++c) s += We[d * 256 + hh * 64 + c] * ae[hh * 64 + c];
    v[d * 4 + hh] = s;
}

// ---------------- edge alpha (leaky-relu) + segment max ----------------
__global__ void edge_alpha_k(const int* __restrict__ src, const int* __restrict__ dst,
                             const float* __restrict__ ea, const float* __restrict__ lea,
                             const float* __restrict__ v, const float* __restrict__ als,
                             const float* __restrict__ ald, float* __restrict__ alpha,
                             float* __restrict__ mbuf, int E, int N) {
    int t = blockIdx.x * blockDim.x + threadIdx.x;
    if (t >= (E + N) * 4) return;
    int e = t >> 2, hh = t & 3;
    int s_, d_; const float* ep;
    if (e < E) { s_ = src[e]; d_ = dst[e]; ep = ea + (long)e * 16; }
    else       { s_ = d_ = e - E;          ep = lea + (long)(e - E) * 16; }
    float ale = 0.f;
#pragma unroll
    for (int d = 0; d < 16; ++d) ale += ep[d] * v[d * 4 + hh];
    float a = als[s_ * 4 + hh] + ald[d_ * 4 + hh] + ale;
    a = a >= 0.f ? a : 0.2f * a;
    alpha[t] = a;
    atomicMax(&mbuf[d_ * 4 + hh], a);
}

// ---------------- exp(alpha - max) + segment sum ----------------
__global__ void edge_exp_k(const int* __restrict__ dst, float* __restrict__ alpha,
                           const float* __restrict__ mbuf, float* __restrict__ sbuf,
                           int E, int N) {
    int t = blockIdx.x * blockDim.x + threadIdx.x;
    if (t >= (E + N) * 4) return;
    int e = t >> 2, hh = t & 3;
    int d_ = (e < E) ? dst[e] : e - E;
    float ex = __expf(alpha[t] - mbuf[d_ * 4 + hh]);
    alpha[t] = ex;
    atomicAdd(&sbuf[d_ * 4 + hh], ex);
}

// ---------------- normalize to attention coefficients ----------------
__global__ void edge_att_k(const int* __restrict__ dst, float* __restrict__ alpha,
                           const float* __restrict__ sbuf, int E, int N) {
    int t = blockIdx.x * blockDim.x + threadIdx.x;
    if (t >= (E + N) * 4) return;
    int e = t >> 2, hh = t & 3;
    int d_ = (e < E) ? dst[e] : e - E;
    alpha[t] = alpha[t] / (sbuf[d_ * 4 + hh] + 1e-16f);
}

// ---------------- scatter-aggregate: out[dst] += h[src] * att ----------------
__global__ void edge_scatter_k(const int* __restrict__ src, const int* __restrict__ dst,
                               const float* __restrict__ att, const float* __restrict__ h,
                               float* __restrict__ out, int E, int N) {
    long t = (long)blockIdx.x * blockDim.x + threadIdx.x;
    if (t >= (long)(E + N) * 64) return;
    int e = (int)(t >> 6), c = (int)(t & 63);
    int s_, d_;
    if (e < E) { s_ = src[e]; d_ = dst[e]; } else { s_ = d_ = e - E; }
    const float* hp = h + (long)s_ * 256 + c;
    float*       op = out + (long)d_ * 256 + c;
#pragma unroll
    for (int hh = 0; hh < 4; ++hh)
        atomicAdd(op + hh * 64, hp[hh * 64] * att[e * 4 + hh]);
}

// ---------------- bias + relu -> bf16 (next layer's GEMM input) ----------------
__global__ void bias_relu_bf16_k(const float* __restrict__ in, const float* __restrict__ b,
                                 __bf16* __restrict__ out, int N) {
    long t = (long)blockIdx.x * blockDim.x + threadIdx.x;
    if (t >= (long)N * 256) return;
    float v = in[t] + b[t & 255];
    out[t] = (__bf16)(v > 0.f ? v : 0.f);
}

// ---------------- head mean + bias (layer 3) ----------------
__global__ void head_mean_k(const float* __restrict__ in, const float* __restrict__ b3,
                            float* __restrict__ fin, int N) {
    int t = blockIdx.x * blockDim.x + threadIdx.x;
    if (t >= N * 64) return;
    int n = t >> 6, c = t & 63;
    const float* p = in + (long)n * 256 + c;
    fin[t] = 0.25f * (p[0] + p[64] + p[128] + p[192]) + b3[c];
}

// ---------------- node mean pool of final features ----------------
__global__ void npool_k(const float* __restrict__ fin, const int* __restrict__ batch,
                        float* __restrict__ psum, int N) {
    int t = blockIdx.x * blockDim.x + threadIdx.x;
    if (t >= N * 64) return;
    int n = t >> 6, c = t & 63;
    atomicAdd(&psum[batch[n] * 64 + c], fin[t]);
}

// ---------------- tiny per-graph MLP head ----------------
__global__ void mlp_k(const float* __restrict__ pooled, const float* __restrict__ gt,
                      const float* __restrict__ mw1, const float* __restrict__ mb1,
                      const float* __restrict__ mw2, const float* __restrict__ mb2,
                      const float* __restrict__ mw3, const float* __restrict__ mb3,
                      float* __restrict__ out) {
    int g = blockIdx.x * blockDim.x + threadIdx.x;
    if (g >= NG) return;
    float z0[80];
    for (int i = 0; i < 64; ++i) z0[i] = pooled[g * 64 + i];
    for (int i = 0; i < 16; ++i) z0[64 + i] = gt[g * 16 + i];
    float z1[64];
    for (int j = 0; j < 64; ++j) {
        float s = mb1[j];
        for (int i = 0; i < 80; ++i) s += z0[i] * mw1[i * 64 + j];
        z1[j] = s > 0.f ? s : 0.f;
    }
    float z2[32];
    for (int j = 0; j < 32; ++j) {
        float s = mb2[j];
        for (int i = 0; i < 64; ++i) s += z1[i] * mw2[i * 32 + j];
        z2[j] = s > 0.f ? s : 0.f;
    }
    for (int j = 0; j < 10; ++j) {
        float s = mb3[j];
        for (int i = 0; i < 32; ++i) s += z2[i] * mw3[i * 10 + j];
        out[g * 10 + j] = s;
    }
}

static inline int cdiv(long a, long b) { return (int)((a + b - 1) / b); }

extern "C" void kernel_launch(void* const* d_in, const int* in_sizes, int n_in,
                              void* d_out, int out_size, void* d_ws, size_t ws_size,
                              hipStream_t stream) {
    const int N = NNODE, E = NEDGE, G = NG;
    const int Ea = E + N;

    const float* x     = (const float*)d_in[0];
    const int*   src   = (const int*)d_in[1];
    const int*   dst   = src + E;
    const float* ea    = (const float*)d_in[2];
    const int*   batch = (const int*)d_in[3];
    const float* w[3]  = { (const float*)d_in[4],  (const float*)d_in[9],  (const float*)d_in[14] };
    const float* as_[3]= { (const float*)d_in[5],  (const float*)d_in[10], (const float*)d_in[15] };
    const float* ad_[3]= { (const float*)d_in[6],  (const float*)d_in[11], (const float*)d_in[16] };
    const float* we_[3]= { (const float*)d_in[7],  (const float*)d_in[12], (const float*)d_in[17] };
    const float* ae_[3]= { (const float*)d_in[8],  (const float*)d_in[13], (const float*)d_in[18] };
    const float* bias[3]={ (const float*)d_in[19], (const float*)d_in[20], (const float*)d_in[21] };
    const float* mw1 = (const float*)d_in[22]; const float* mb1 = (const float*)d_in[23];
    const float* mw2 = (const float*)d_in[24]; const float* mb2 = (const float*)d_in[25];
    const float* mw3 = (const float*)d_in[26]; const float* mb3 = (const float*)d_in[27];

    // ---- workspace layout ----
    float* ws = (float*)d_ws;
    float* cnt    = ws;                       // N
    float* sumea  = cnt + N;                  // N*16 (becomes loop_ea in place)
    float* bufA   = sumea + N * 16;           // N*256  (GEMM output h)
    float* bufB   = bufA + N * 256;           // N*256  (aggregation out)
    float* als    = bufB + N * 256;           // N*4
    float* ald    = als + N * 4;              // N*4
    float* mbuf   = ald + N * 4;              // N*4
    float* sbuf   = mbuf + N * 4;             // N*4
    float* alpha  = sbuf + N * 4;             // Ea*4
    float* ve     = alpha + (long)Ea * 4;     // 64
    float* gsum   = ve + 64;                  // G*16
    float* gcnt   = gsum + G * 16;            // G
    float* gt     = gcnt + G;                 // G*16
    float* psum   = gt + G * 16;              // G*64
    float* pooled = psum + G * 64;            // G*64
    __bf16* xbf   = (__bf16*)(pooled + G * 64);   // N*64 bf16 (layer-1 input)
    __bf16* abf   = xbf + (long)N * 64;           // N*256 bf16 (layer-2/3 input)
    __bf16* wtb   = abf + (long)N * 256;          // 256*256 bf16 (transposed weights)

    // ---- self-loop edge attrs + graph-feature pooling ----
    fill_k<<<cdiv((long)N + N * 16, 256), 256, 0, stream>>>(cnt, 0.f, (long)N + N * 16);
    edge_stats_k<<<cdiv((long)E * 16, 256), 256, 0, stream>>>(dst, ea, cnt, sumea, E);
    loop_ea_k<<<cdiv((long)N * 16, 256), 256, 0, stream>>>(cnt, sumea, N);

    fill_k<<<1, 256, 0, stream>>>(gsum, 0.f, (long)G * 16 + G);
    gstat_k<<<cdiv((long)N * 16, 256), 256, 0, stream>>>(x, batch, gsum, gcnt, N);
    gdiv_k<<<cdiv((long)G * 16, 256), 256, 0, stream>>>(gsum, gcnt, gt, G, 16);

    pack_x_bf16_k<<<cdiv((long)N * 64, 256), 256, 0, stream>>>(x, xbf, N);

    // ---- 3 GAT layers ----
    const __bf16* Ain = xbf;
    int K = 64;
    float* hbuf = bufA;
    float* obuf = bufB;
    const int gemmWaves = (N / 16) * 4;          // 5000 waves, 8/block -> 625 full blocks

    for (int l = 0; l < 3; ++l) {
        wt_bf16_k<<<cdiv((long)K * 256, 256), 256, 0, stream>>>(w[l], wtb, K);
        gemm_bf16t_wmma<<<gemmWaves / 8, 256, 0, stream>>>(Ain, wtb, hbuf, N, K);
        edge_vec_k<<<1, 64, 0, stream>>>(we_[l], ae_[l], ve);
        node_alpha_k<<<cdiv((long)N * 4, 256), 256, 0, stream>>>(hbuf, as_[l], ad_[l], als, ald, N);
        fill_k<<<cdiv((long)N * 4, 256), 256, 0, stream>>>(mbuf, -1e30f, (long)N * 4);
        fill_k<<<cdiv((long)N * 4, 256), 256, 0, stream>>>(sbuf, 0.f, (long)N * 4);
        edge_alpha_k<<<cdiv((long)Ea * 4, 256), 256, 0, stream>>>(src, dst, ea, sumea, ve,
                                                                  als, ald, alpha, mbuf, E, N);
        edge_exp_k<<<cdiv((long)Ea * 4, 256), 256, 0, stream>>>(dst, alpha, mbuf, sbuf, E, N);
        edge_att_k<<<cdiv((long)Ea * 4, 256), 256, 0, stream>>>(dst, alpha, sbuf, E, N);
        fill_k<<<cdiv((long)N * 256, 256), 256, 0, stream>>>(obuf, 0.f, (long)N * 256);
        edge_scatter_k<<<cdiv((long)Ea * 64, 256), 256, 0, stream>>>(src, dst, alpha, hbuf, obuf, E, N);

        if (l < 2) {
            bias_relu_bf16_k<<<cdiv((long)N * 256, 256), 256, 0, stream>>>(obuf, bias[l], abf, N);
            Ain = abf; K = 256;
        } else {
            head_mean_k<<<cdiv((long)N * 64, 256), 256, 0, stream>>>(obuf, bias[2], hbuf, N);
        }
    }
    float* fin = hbuf;   // N x 64 final node features

    // ---- node mean pool + MLP head ----
    fill_k<<<cdiv((long)G * 64, 256), 256, 0, stream>>>(psum, 0.f, (long)G * 64);
    npool_k<<<cdiv((long)N * 64, 256), 256, 0, stream>>>(fin, batch, psum, N);
    gdiv_k<<<cdiv((long)G * 64, 256), 256, 0, stream>>>(psum, gcnt, pooled, G, 64);
    mlp_k<<<1, 64, 0, stream>>>(pooled, gt, mw1, mb1, mw2, mb2, mw3, mb3, (float*)d_out);
}
